// GNNModel_71176198029767
// MI455X (gfx1250) — matrix operand order repaired and checked by reading, not compiled, not run on hardware
//
#include <hip/hip_runtime.h>

#define NN 100000   // nodes
#define NB 300000   // bonds (a2a edges)
#define NA 600000   // angles (bab edges)
#define NG 5000     // graphs

typedef __attribute__((ext_vector_type(16))) __bf16 v16bf;
typedef __attribute__((ext_vector_type(8)))  float  v8f;

union FragBF {
  uint4 u[2];
  v16bf v;
};

__device__ __forceinline__ unsigned short f2bf(float f) {
  unsigned int u = __float_as_uint(f);
  u += 0x7FFFu + ((u >> 16) & 1u);   // round-to-nearest-even
  return (unsigned short)(u >> 16);
}

// ---------------------------------------------------------------------------
// Y[M,128] = act( X_bf16[M,K] @ W_f32[K,128] + bias ), W swizzled to LDS bf16.
// 256 threads = 8 waves; wave computes a 16x128 row-slab via 8 WMMA tiles.
// A layout (16-bit, 16x32): lanes 0-15 rows M=0..15 hold K = {0..7,16..23},
// lanes 16-31 same rows hold K = {8..15,24..31}  -> two 16B loads per lane.
// B layout (16-bit, 32x16): lane = n + 16*((k>>3)&1), same K order as A per lane.
// ---------------------------------------------------------------------------
__global__ __launch_bounds__(256) void gemm_bf16_kernel(
    const unsigned short* __restrict__ X, const float* __restrict__ W,
    const float* __restrict__ bias, float* __restrict__ Yf,
    unsigned short* __restrict__ Ybf, int M, int K, int relu)
{
  extern __shared__ unsigned short sW[];
  const int tid = threadIdx.x;
  const int nkt = K >> 5;                       // K tiles of 32
  // Convert + swizzle W into B-fragment order: [ntile][kt][lane][t(0..15)]
  for (int idx = tid; idx < K * 128; idx += 256) {
    int k = idx >> 7, n = idx & 127;
    int kt = k >> 5, kin = k & 31;
    int lane = (n & 15) | (((kin >> 3) & 1) << 4);
    int t = (kin & 7) | ((kin >> 4) << 3);
    sW[((((n >> 4) * nkt + kt) << 5 | lane) << 4) | t] = f2bf(W[idx]);
  }
  __syncthreads();

  const int lane = tid & 31;
  const int wave = tid >> 5;
  const int base = blockIdx.x * 128 + wave * 16;
  int m0 = base + (lane & 15);
  if (m0 > M - 1) m0 = M - 1;                   // clamp OOB rows (stores guarded)
  const int khalf = (lane >> 4) << 3;           // 0 or 8

  v8f acc[8] = {};
  for (int kt = 0; kt < nkt; ++kt) {
    FragBF a;
    const unsigned short* xr = X + (size_t)m0 * K + (kt << 5) + khalf;
    a.u[0] = *(const uint4*)(xr);
    a.u[1] = *(const uint4*)(xr + 16);
#pragma unroll
    for (int nt = 0; nt < 8; ++nt) {
      FragBF b;
      const uint4* bp = (const uint4*)(sW + ((((nt * nkt + kt) << 5) | lane) << 4));
      b.u[0] = bp[0];
      b.u[1] = bp[1];
      acc[nt] = __builtin_amdgcn_wmma_f32_16x16x32_bf16(
          false, a.v, false, b.v, (short)0, acc[nt], false, false);
    }
  }

  const int n0 = lane & 15;
  const int mo = (lane >> 4) << 3;
#pragma unroll
  for (int nt = 0; nt < 8; ++nt) {
    int n = (nt << 4) | n0;
    float bv = bias ? bias[n] : 0.f;
#pragma unroll
    for (int j = 0; j < 8; ++j) {
      int m = base + mo + j;
      if (m < M) {
        float v = acc[nt][j] + bv;
        if (relu) v = fmaxf(v, 0.f);
        Yf[(size_t)m * 128 + n] = v;
        if (Ybf) Ybf[(size_t)m * 128 + n] = f2bf(v);
      }
    }
  }
}

// angle_h = exp(-10*(ang - c)^2) @ rbf_W + rbf_b  -> bf16  [NA,128]
__global__ __launch_bounds__(256) void rbf_kernel(
    const float* __restrict__ angle, const float* __restrict__ rbfW,
    const float* __restrict__ rbfB, unsigned short* __restrict__ out, int E)
{
  __shared__ float sw[32 * 128];
  __shared__ float sb[128];
  __shared__ float sr[2][32];
  int tid = threadIdx.x;
  for (int i = tid; i < 32 * 128; i += 256) sw[i] = rbfW[i];
  if (tid < 128) sb[tid] = rbfB[tid];
  int le = tid >> 7, d = tid & 127;
  long e = (long)blockIdx.x * 2 + le;
  float ang = (e < E) ? angle[e] : 0.f;
  if (d < 32) { float t = ang - 0.1f * (float)d; sr[le][d] = __expf(-10.f * t * t); }
  __syncthreads();
  float acc = sb[d];
#pragma unroll
  for (int c = 0; c < 32; ++c) acc = fmaf(sr[le][c], sw[c * 128 + d], acc);
  if (e < E) out[e * 128 + d] = f2bf(acc);
}

__global__ __launch_bounds__(256) void cvt_kernel(
    const float* __restrict__ in, unsigned short* __restrict__ out, long n)
{
  long i = (long)blockIdx.x * 256 + threadIdx.x;
  if (i < n) out[i] = f2bf(in[i]);
}

// per-row head dots: out[row,h] = sum_dh h[row,h*32+dh]*attn[h,dh]; wave/row
__global__ __launch_bounds__(256) void scores_kernel(
    const float* __restrict__ h, const float* __restrict__ attnA,
    const float* __restrict__ attnB, float* __restrict__ outA,
    float* __restrict__ outB, int N)
{
  int lane = threadIdx.x & 31;
  int row = blockIdx.x * 8 + (threadIdx.x >> 5);
  if (row >= N) return;
  const float* hr = h + (size_t)row * 128;
#pragma unroll
  for (int hd = 0; hd < 4; ++hd) {
    float x = hr[hd * 32 + lane];
    float va = x * attnA[hd * 32 + lane];
    float vb = attnB ? x * attnB[hd * 32 + lane] : 0.f;
#pragma unroll
    for (int off = 16; off > 0; off >>= 1) {
      va += __shfl_xor(va, off, 32);
      vb += __shfl_xor(vb, off, 32);
    }
    if (lane == 0) {
      outA[row * 4 + hd] = va;
      if (outB) outB[row * 4 + hd] = vb;
    }
  }
}

__device__ __forceinline__ void atomicMaxF(float* addr, float v) {
  if (v >= 0.f) atomicMax((int*)addr, __float_as_int(v));
  else          atomicMin((unsigned int*)addr, __float_as_uint(v));
}

__global__ __launch_bounds__(256) void fill_kernel(float* __restrict__ p, float v, long n) {
  long i = (long)blockIdx.x * 256 + threadIdx.x;
  if (i < n) p[i] = v;
}

__global__ __launch_bounds__(256) void edge_max_kernel(
    const int* __restrict__ src, const int* __restrict__ dst,
    const float* __restrict__ es, const float* __restrict__ ed,
    const float* __restrict__ er, float* __restrict__ ebuf,
    float* __restrict__ mseg, long E)
{
  long i = (long)blockIdx.x * 256 + threadIdx.x;
  if (i >= E * 4) return;
  long e = i >> 2; int hd = (int)(i & 3);
  int s = src[e], d = dst[e];
  float v = es[s * 4 + hd] + ed[d * 4 + hd] + er[i];
  v = (v > 0.f) ? v : 0.2f * v;                 // leaky relu
  ebuf[i] = v;
  atomicMaxF(&mseg[d * 4 + hd], v);
}

__global__ __launch_bounds__(256) void edge_exp_kernel(
    const int* __restrict__ dst, const float* __restrict__ ebuf,
    const float* __restrict__ mseg, float* __restrict__ exbuf,
    float* __restrict__ sseg, long E)
{
  long i = (long)blockIdx.x * 256 + threadIdx.x;
  if (i >= E * 4) return;
  long e = i >> 2; int hd = (int)(i & 3);
  int d = dst[e];
  float ex = __expf(ebuf[i] - mseg[d * 4 + hd]);
  exbuf[i] = ex;
  atomicAdd(&sseg[d * 4 + hd], ex);
}

// agg[dst,:] += (ex/(s+eps)) * (h[src,:] + r[e,:]); 128 threads per edge
__global__ __launch_bounds__(256) void edge_agg_kernel(
    const int* __restrict__ src, const int* __restrict__ dst,
    const float* __restrict__ exbuf, const float* __restrict__ sseg,
    const float* __restrict__ h, const float* __restrict__ r,
    float* __restrict__ agg, long E)
{
  long e = (long)blockIdx.x * 2 + (threadIdx.x >> 7);
  int d = threadIdx.x & 127;
  if (e >= E) return;
  int s = src[e], dd = dst[e];
  int hd = d >> 5;
  float coef = exbuf[e * 4 + hd] / (sseg[dd * 4 + hd] + 1e-9f);
  float val = coef * (h[(size_t)s * 128 + d] + r[e * 128 + d]);
  atomicAdd(&agg[(size_t)dd * 128 + d], val);
}

// out = relu(0.95*agg + 0.05*ent) in f32 and bf16
__global__ __launch_bounds__(256) void residual_kernel(
    const float* __restrict__ agg, const float* __restrict__ ent,
    float* __restrict__ outf, unsigned short* __restrict__ outbf, long n)
{
  long i = (long)blockIdx.x * 256 + threadIdx.x;
  if (i >= n) return;
  float v = fmaxf(0.95f * agg[i] + 0.05f * ent[i], 0.f);
  outf[i] = v;
  outbf[i] = f2bf(v);
}

// tri[e, 0:128]=g[src], [128:256]=bond[e], [256:384]=g[dst]   (bf16)
__global__ __launch_bounds__(256) void tri_gather_kernel(
    const int* __restrict__ src, const int* __restrict__ dst,
    const unsigned short* __restrict__ gbf, const unsigned short* __restrict__ bondbf,
    unsigned short* __restrict__ tri, long E)
{
  long i = (long)blockIdx.x * 256 + threadIdx.x;
  if (i >= E * 384) return;
  long e = i / 384;
  int d = (int)(i - e * 384);
  unsigned short v;
  if (d < 128)      v = gbf[(size_t)src[e] * 128 + d];
  else if (d < 256) v = bondbf[e * 128 + (d - 128)];
  else              v = gbf[(size_t)dst[e] * 128 + (d - 256)];
  tri[i] = v;
}

__global__ __launch_bounds__(256) void count_kernel(
    const int* __restrict__ gid, float* __restrict__ cnt, int n)
{
  int i = blockIdx.x * 256 + threadIdx.x;
  if (i < n) atomicAdd(&cnt[gid[i]], 1.f);
}

__global__ __launch_bounds__(256) void pool_sum_kernel(
    const float* __restrict__ h, const int* __restrict__ gid,
    float* __restrict__ out, long nrows)
{
  long i = (long)blockIdx.x * 256 + threadIdx.x;
  if (i >= nrows * 128) return;
  long row = i >> 7; int d = (int)(i & 127);
  atomicAdd(&out[(size_t)gid[row] * 128 + d], h[i]);
}

__global__ __launch_bounds__(256) void pool_div_kernel(
    float* __restrict__ out, const float* __restrict__ cnt, int G)
{
  int i = blockIdx.x * 256 + threadIdx.x;
  if (i < G * 128) out[i] /= fmaxf(cnt[i >> 7], 1.f);
}

// ---------------------------------------------------------------------------
extern "C" void kernel_launch(void* const* d_in, const int* in_sizes, int n_in,
                              void* d_out, int out_size, void* d_ws, size_t ws_size,
                              hipStream_t stream) {
  (void)in_sizes; (void)n_in; (void)out_size; (void)ws_size;
  // d_in layout = jax tree_leaves order (dict keys sorted):
  // 0 node_h, 1 bond_h, 2 bond_angle,
  // 3..14  block0: W_ent1,W_rel1,attn_dst1,attn_rel1,attn_src1,
  //                W_ent2,W_rel2,attn_dst2,attn_rel2,attn_src2, tri_W, tri_b
  // 15..26 block1 (same), 27 rbf_W, 28 rbf_b,
  // 29 a2a_src, 30 a2a_dst, 31 bab_src, 32 bab_dst, 33 node2graph, 34 bond2graph
  const float* node_h     = (const float*)d_in[0];
  const float* bond_h     = (const float*)d_in[1];
  const float* bond_angle = (const float*)d_in[2];
  const float* rbf_W      = (const float*)d_in[27];
  const float* rbf_b      = (const float*)d_in[28];
  const int* a2a_src = (const int*)d_in[29];
  const int* a2a_dst = (const int*)d_in[30];
  const int* bab_src = (const int*)d_in[31];
  const int* bab_dst = (const int*)d_in[32];
  const int* node2g  = (const int*)d_in[33];
  const int* bond2g  = (const int*)d_in[34];
  float* out = (float*)d_out;

  // ---- workspace bump allocator ----
  char* w = (char*)d_ws;
  size_t off = 0;
  auto alloc = [&](size_t bytes) -> void* {
    void* p = w + off;
    off = (off + bytes + 255) & ~(size_t)255;
    return p;
  };
  unsigned short* angle_bf  = (unsigned short*)alloc((size_t)NA * 128 * 2);
  unsigned short* node_bf   = (unsigned short*)alloc((size_t)NN * 128 * 2);
  unsigned short* bond_bf   = (unsigned short*)alloc((size_t)NB * 128 * 2);
  float*          node_f    = (float*)alloc((size_t)NN * 128 * 4);
  float*          bond_f    = (float*)alloc((size_t)NB * 128 * 4);
  float*          h_f       = (float*)alloc((size_t)NB * 128 * 4);
  float*          r_f       = (float*)alloc((size_t)NA * 128 * 4);
  float*          agg_f     = (float*)alloc((size_t)NB * 128 * 4);
  unsigned short* tri_bf    = (unsigned short*)alloc((size_t)NB * 384 * 2);
  float*          triout_f  = (float*)alloc((size_t)NB * 128 * 4);
  unsigned short* triout_bf = (unsigned short*)alloc((size_t)NB * 128 * 2);
  float* es    = (float*)alloc((size_t)NB * 4 * 4);
  float* ed    = (float*)alloc((size_t)NB * 4 * 4);
  float* er    = (float*)alloc((size_t)NA * 4 * 4);
  float* ebuf  = (float*)alloc((size_t)NA * 4 * 4);
  float* exbuf = (float*)alloc((size_t)NA * 4 * 4);
  float* mseg  = (float*)alloc((size_t)NB * 4 * 4);
  float* sseg  = (float*)alloc((size_t)NB * 4 * 4);
  float* cntN  = (float*)alloc((size_t)NG * 4);
  float* cntB  = (float*)alloc((size_t)NG * 4);

  auto blk1d = [](long n) { return dim3((unsigned)((n + 255) / 256)); };
  auto gemm = [&](const unsigned short* X, const float* W_, const float* bias,
                  float* Yf, unsigned short* Ybf, int M, int K, int relu) {
    gemm_bf16_kernel<<<dim3((M + 127) / 128), dim3(256), (size_t)K * 128 * 2, stream>>>(
        X, W_, bias, Yf, Ybf, M, K, relu);
  };

  // ---- setup: zero pooled outputs, counts; rbf; bf16 copies of inputs ----
  hipMemsetAsync(d_out, 0, (size_t)4 * NG * 128 * 4, stream);   // 4 pooled blocks
  hipMemsetAsync(cntN, 0, (size_t)NG * 4, stream);
  hipMemsetAsync(cntB, 0, (size_t)NG * 4, stream);
  count_kernel<<<blk1d(NN), 256, 0, stream>>>(node2g, cntN, NN);
  count_kernel<<<blk1d(NB), 256, 0, stream>>>(bond2g, cntB, NB);
  rbf_kernel<<<dim3((NA + 1) / 2), 256, 0, stream>>>(bond_angle, rbf_W, rbf_b, angle_bf, NA);
  cvt_kernel<<<blk1d((long)NN * 128), 256, 0, stream>>>(node_h, node_bf, (long)NN * 128);
  cvt_kernel<<<blk1d((long)NB * 128), 256, 0, stream>>>(bond_h, bond_bf, (long)NB * 128);

  for (int blk = 0; blk < 2; ++blk) {
    int pb = 3 + blk * 12;
    const float* W_ent1    = (const float*)d_in[pb + 0];
    const float* W_rel1    = (const float*)d_in[pb + 1];
    const float* attn_dst1 = (const float*)d_in[pb + 2];
    const float* attn_rel1 = (const float*)d_in[pb + 3];
    const float* attn_src1 = (const float*)d_in[pb + 4];
    const float* W_ent2    = (const float*)d_in[pb + 5];
    const float* W_rel2    = (const float*)d_in[pb + 6];
    const float* attn_dst2 = (const float*)d_in[pb + 7];
    const float* attn_rel2 = (const float*)d_in[pb + 8];
    const float* attn_src2 = (const float*)d_in[pb + 9];
    const float* tri_W     = (const float*)d_in[pb + 10];
    const float* tri_b     = (const float*)d_in[pb + 11];

    // ---- gnn1 (nodes, a2a edges) ----
    gemm(node_bf, W_ent1, nullptr, h_f, nullptr, NN, 128, 0);
    gemm(bond_bf, W_rel1, nullptr, r_f, nullptr, NB, 128, 0);
    scores_kernel<<<dim3((NN + 7) / 8), 256, 0, stream>>>(h_f, attn_src1, attn_dst1, es, ed, NN);
    scores_kernel<<<dim3((NB + 7) / 8), 256, 0, stream>>>(r_f, attn_rel1, nullptr, er, nullptr, NB);
    fill_kernel<<<blk1d((long)NN * 4), 256, 0, stream>>>(mseg, -1e38f, (long)NN * 4);
    hipMemsetAsync(sseg, 0, (size_t)NN * 4 * 4, stream);
    edge_max_kernel<<<blk1d((long)NB * 4), 256, 0, stream>>>(a2a_src, a2a_dst, es, ed, er, ebuf, mseg, NB);
    edge_exp_kernel<<<blk1d((long)NB * 4), 256, 0, stream>>>(a2a_dst, ebuf, mseg, exbuf, sseg, NB);
    hipMemsetAsync(agg_f, 0, (size_t)NN * 128 * 4, stream);
    edge_agg_kernel<<<dim3((NB + 1) / 2), 256, 0, stream>>>(a2a_src, a2a_dst, exbuf, sseg, h_f, r_f, agg_f, NB);
    const float* ent1 = (blk == 0) ? node_h : node_f;
    residual_kernel<<<blk1d((long)NN * 128), 256, 0, stream>>>(agg_f, ent1, node_f, node_bf, (long)NN * 128);
    pool_sum_kernel<<<blk1d((long)NN * 128), 256, 0, stream>>>(node_f, node2g, out + (size_t)blk * NG * 128, NN);

    // ---- tri MLP (gathered 384-wide edge features) ----
    tri_gather_kernel<<<blk1d((long)NB * 384), 256, 0, stream>>>(a2a_src, a2a_dst, node_bf, bond_bf, tri_bf, NB);
    gemm(tri_bf, tri_W, tri_b, triout_f, triout_bf, NB, 384, 1);

    // ---- gnn2 (bonds, bab edges) ----
    gemm(triout_bf, W_ent2, nullptr, h_f, nullptr, NB, 128, 0);
    gemm(angle_bf, W_rel2, nullptr, r_f, nullptr, NA, 128, 0);
    scores_kernel<<<dim3((NB + 7) / 8), 256, 0, stream>>>(h_f, attn_src2, attn_dst2, es, ed, NB);
    scores_kernel<<<dim3((NA + 7) / 8), 256, 0, stream>>>(r_f, attn_rel2, nullptr, er, nullptr, NA);
    fill_kernel<<<blk1d((long)NB * 4), 256, 0, stream>>>(mseg, -1e38f, (long)NB * 4);
    hipMemsetAsync(sseg, 0, (size_t)NB * 4 * 4, stream);
    edge_max_kernel<<<blk1d((long)NA * 4), 256, 0, stream>>>(bab_src, bab_dst, es, ed, er, ebuf, mseg, NA);
    edge_exp_kernel<<<blk1d((long)NA * 4), 256, 0, stream>>>(bab_dst, ebuf, mseg, exbuf, sseg, NA);
    hipMemsetAsync(agg_f, 0, (size_t)NB * 128 * 4, stream);
    edge_agg_kernel<<<dim3((NA + 1) / 2), 256, 0, stream>>>(bab_src, bab_dst, exbuf, sseg, h_f, r_f, agg_f, NA);
    float* g2f = (blk == 1) ? (out + (size_t)4 * NG * 128) : bond_f;   // bond_rep on last block
    residual_kernel<<<blk1d((long)NB * 128), 256, 0, stream>>>(agg_f, triout_f, g2f, bond_bf, (long)NB * 128);
    pool_sum_kernel<<<blk1d((long)NB * 128), 256, 0, stream>>>(g2f, bond2g, out + (size_t)(2 + blk) * NG * 128, NB);
  }

  // ---- mean = sum / count ----
  pool_div_kernel<<<blk1d((long)NG * 128), 256, 0, stream>>>(out + 0 * (size_t)NG * 128, cntN, NG);
  pool_div_kernel<<<blk1d((long)NG * 128), 256, 0, stream>>>(out + 1 * (size_t)NG * 128, cntN, NG);
  pool_div_kernel<<<blk1d((long)NG * 128), 256, 0, stream>>>(out + 2 * (size_t)NG * 128, cntB, NG);
  pool_div_kernel<<<blk1d((long)NG * 128), 256, 0, stream>>>(out + 3 * (size_t)NG * 128, cntB, NG);
}
